// MambaS6Layer_7043746365499
// MI455X (gfx1250) — matrix operand order repaired and checked by reading, compiled, not run on hardware
//
#include <hip/hip_runtime.h>

// ---------------------------------------------------------------------------
// Mamba S6 forward for MI455X (gfx1250, wave32, WMMA bf16 16x16x32)
//   B=4, S=2048, D_MODEL=1024, D_INNER=2048, D_STATE=16, D_CONV=3
// GEMMs: f32 operands converted to bf16 while staging into double-buffered
// LDS tiles; f32 WMMA accumulate (all operands fit in the 192MB L2).
// Scan: chunked two-level associative scan (16 chunks x 128 steps) for 16x
// the wave-level parallelism of a naive sequential scan.
// ---------------------------------------------------------------------------

typedef __attribute__((ext_vector_type(16))) __bf16 bf16x16;
typedef __attribute__((ext_vector_type(8)))  __bf16 bf16x8;
typedef __attribute__((ext_vector_type(8)))  float  f32x8;

#define GEMM_BM 128
#define GEMM_BN 128
#define GEMM_BK 32
#define LDS_STRIDE 40   // 32 bf16 + 8 pad = 80 bytes/row: 16B aligned, conflict-free

#define SCAN_CHUNK  128
#define SCAN_NCHUNK 16  // S / SCAN_CHUNK

__device__ __forceinline__ float softplus_f(float t) {
    return (t > 20.0f) ? t : log1pf(__expf(t));
}

// C[M,N] = A[M,K] @ B[K,N]; A,B f32 in global, C f32.
// M % 128 == 0, K % 32 == 0; N guarded (block-uniform fast path).
__global__ __launch_bounds__(256) void gemm_bf16_wmma(
    const float* __restrict__ A, const float* __restrict__ Bm,
    float* __restrict__ C, int M, int N, int K, int lda, int ldb, int ldc)
{
    __shared__ __bf16 As[2][GEMM_BM * LDS_STRIDE];
    __shared__ __bf16 Bs[2][GEMM_BN * LDS_STRIDE];

    const int tid  = threadIdx.x;
    const int lane = tid & 31;
    const int wid  = tid >> 5;        // 8 waves
    const int wm   = wid >> 2;        // 0..1  -> 64-row strip
    const int wn   = wid & 3;         // 0..3  -> 32-col strip
    const int lrow = lane & 15;
    const int half = lane >> 4;       // 0/1: lane group selector
    const int mBase = blockIdx.y * GEMM_BM;
    const int nBase = blockIdx.x * GEMM_BN;
    const bool interior = (nBase + GEMM_BN) <= N;   // block-uniform

    int amRow[4], akCol[4], bkRow[4], bnCol[4];
#pragma unroll
    for (int it = 0; it < 4; ++it) {
        int idx = tid + it * 256;       // 0..1023 float4 slots
        amRow[it] = idx >> 3;           // 0..127
        akCol[it] = (idx & 7) << 2;     // 0,4,...,28
        bkRow[it] = idx >> 5;           // 0..31
        bnCol[it] = (idx & 31) << 2;    // 0..124
    }

    float4 aReg[4], bReg[4];

    auto loadTile = [&](int k0) {
#pragma unroll
        for (int it = 0; it < 4; ++it)
            aReg[it] = *reinterpret_cast<const float4*>(
                A + (size_t)(mBase + amRow[it]) * lda + (k0 + akCol[it]));
        if (interior) {
#pragma unroll
            for (int it = 0; it < 4; ++it)
                bReg[it] = *reinterpret_cast<const float4*>(
                    Bm + (size_t)(k0 + bkRow[it]) * ldb + (nBase + bnCol[it]));
        } else {
#pragma unroll
            for (int it = 0; it < 4; ++it) {
                const float* bp = Bm + (size_t)(k0 + bkRow[it]) * ldb;
                int gn = nBase + bnCol[it];
                bReg[it].x = (gn + 0 < N) ? bp[gn + 0] : 0.0f;
                bReg[it].y = (gn + 1 < N) ? bp[gn + 1] : 0.0f;
                bReg[it].z = (gn + 2 < N) ? bp[gn + 2] : 0.0f;
                bReg[it].w = (gn + 3 < N) ? bp[gn + 3] : 0.0f;
            }
        }
    };

    auto storeTile = [&](int buf) {
#pragma unroll
        for (int it = 0; it < 4; ++it) {
            __bf16* dst = &As[buf][amRow[it] * LDS_STRIDE + akCol[it]];
            dst[0] = (__bf16)aReg[it].x; dst[1] = (__bf16)aReg[it].y;
            dst[2] = (__bf16)aReg[it].z; dst[3] = (__bf16)aReg[it].w;
        }
#pragma unroll
        for (int it = 0; it < 4; ++it) {
            Bs[buf][(bnCol[it] + 0) * LDS_STRIDE + bkRow[it]] = (__bf16)bReg[it].x;
            Bs[buf][(bnCol[it] + 1) * LDS_STRIDE + bkRow[it]] = (__bf16)bReg[it].y;
            Bs[buf][(bnCol[it] + 2) * LDS_STRIDE + bkRow[it]] = (__bf16)bReg[it].z;
            Bs[buf][(bnCol[it] + 3) * LDS_STRIDE + bkRow[it]] = (__bf16)bReg[it].w;
        }
    };

    f32x8 acc[4][2];
#pragma unroll
    for (int i = 0; i < 4; ++i)
#pragma unroll
        for (int j = 0; j < 2; ++j)
#pragma unroll
            for (int r = 0; r < 8; ++r) acc[i][j][r] = 0.0f;

    const int nk = K / GEMM_BK;
    loadTile(0);
    storeTile(0);
    __syncthreads();

    for (int kt = 0; kt < nk; ++kt) {
        const int buf = kt & 1;
        const bool more = (kt + 1) < nk;
        if (more) loadTile((kt + 1) * GEMM_BK);  // cover latency behind WMMAs

        union Frag { bf16x16 v; bf16x8 h[2]; };
        bf16x16 afrag[4], bfrag[2];
#pragma unroll
        for (int i = 0; i < 4; ++i) {
            const __bf16* p =
                &As[buf][(wm * 64 + i * 16 + lrow) * LDS_STRIDE + half * 8];
            Frag f;
            f.h[0] = *reinterpret_cast<const bf16x8*>(p);        // K 0..7 (+8)
            f.h[1] = *reinterpret_cast<const bf16x8*>(p + 16);   // K 16..23 (+8)
            afrag[i] = f.v;
        }
#pragma unroll
        for (int j = 0; j < 2; ++j) {
            const __bf16* p =
                &Bs[buf][(wn * 32 + j * 16 + lrow) * LDS_STRIDE + half * 8];
            Frag f;
            f.h[0] = *reinterpret_cast<const bf16x8*>(p);
            f.h[1] = *reinterpret_cast<const bf16x8*>(p + 16);
            bfrag[j] = f.v;
        }
#pragma unroll
        for (int i = 0; i < 4; ++i)
#pragma unroll
            for (int j = 0; j < 2; ++j)
                acc[i][j] = __builtin_amdgcn_wmma_f32_16x16x32_bf16(
                    false, afrag[i], false, bfrag[j],
                    (short)0, acc[i][j], false, false);

        if (more) storeTile(buf ^ 1);
        __syncthreads();
    }

#pragma unroll
    for (int i = 0; i < 4; ++i)
#pragma unroll
        for (int j = 0; j < 2; ++j) {
            int row0 = mBase + wm * 64 + i * 16 + half * 8;
            int col  = nBase + wn * 32 + j * 16 + lrow;
            if (col < N) {
#pragma unroll
                for (int r = 0; r < 8; ++r)
                    C[(size_t)(row0 + r) * ldc + col] = acc[i][j][r];
            }
        }
}

// depthwise conv1d (k=3, SAME) + bias + SiLU over u-half of `combined`
__global__ __launch_bounds__(256) void conv_silu_kernel(
    const float* __restrict__ comb, const float* __restrict__ conv_w,
    const float* __restrict__ conv_b, float* __restrict__ u)
{
    const int d = blockIdx.x * 256 + threadIdx.x;   // 0..2047
    const int s = blockIdx.y;                       // 0..2047
    const int b = blockIdx.z;                       // 0..3
    const size_t row = (size_t)b * 2048 + s;
    const float w0 = conv_w[d * 3 + 0];
    const float w1 = conv_w[d * 3 + 1];
    const float w2 = conv_w[d * 3 + 2];
    float xm = (s > 0)    ? comb[(row - 1) * 4096 + d] : 0.0f;
    float xc =              comb[(row    ) * 4096 + d];
    float xp = (s < 2047) ? comb[(row + 1) * 4096 + d] : 0.0f;
    float v = w0 * xm + w1 * xc + w2 * xp + conv_b[d];
    float sig = 1.0f / (1.0f + __expf(-v));
    u[row * 2048 + d] = v * sig;
}

// ---- chunked scan ---------------------------------------------------------
// Summary layout: idx(c,b,n,d) = ((c*4 + b)*16 + n)*2048 + d  (d coalesced)

// Pass 1: per (b,d,chunk): decay product P and local state h (h_in = 0).
// softplus(delta_raw + dt_bias) computed inline.
__global__ __launch_bounds__(256) void scan_pass1(
    const float* __restrict__ draw, const float* __restrict__ dtB,
    const float* __restrict__ u, const float* __restrict__ xdbl,
    const float* __restrict__ A_log,
    float* __restrict__ Pout, float* __restrict__ Hloc)
{
    const int d = blockIdx.x * 256 + threadIdx.x;   // 0..2047
    const int c = blockIdx.y;                       // 0..15
    const int b = blockIdx.z;                       // 0..3
    float a[16], h[16], P[16];
#pragma unroll
    for (int n = 0; n < 16; ++n) {
        a[n] = -__expf(A_log[(size_t)d * 16 + n]);
        h[n] = 0.0f;
        P[n] = 1.0f;
    }
    const float bias = dtB[d];
    const size_t base = (size_t)b * 2048;

    for (int s = c * SCAN_CHUNK; s < (c + 1) * SCAN_CHUNK; ++s) {
        const size_t row = base + s;
        float dt = softplus_f(draw[row * 2048 + d] + bias);
        float ut = u[row * 2048 + d];
        union BC { float4 q[4]; float f[16]; } bv;
        const float4* bp4 = reinterpret_cast<const float4*>(xdbl + row * 2080 + 2048);
#pragma unroll
        for (int q = 0; q < 4; ++q) bv.q[q] = bp4[q];
#pragma unroll
        for (int n = 0; n < 16; ++n) {
            float e = __expf(dt * a[n]);
            h[n] = e * h[n] + (dt * bv.f[n]) * ut;
            P[n] *= e;
        }
    }
    const size_t o = (size_t)((c * 4 + b) * 16) * 2048 + d;
#pragma unroll
    for (int n = 0; n < 16; ++n) {
        Pout[o + (size_t)n * 2048] = P[n];
        Hloc[o + (size_t)n * 2048] = h[n];
    }
}

// Pass 2: per (b,d): 16 sequential chunk combines; write entry state per chunk.
__global__ __launch_bounds__(256) void scan_combine(
    const float* __restrict__ Pin, const float* __restrict__ Hloc,
    float* __restrict__ Hin)
{
    const int d = (blockIdx.x & 7) * 256 + threadIdx.x;   // 0..2047
    const int b = blockIdx.x >> 3;                        // 0..3
    float h[16];
#pragma unroll
    for (int n = 0; n < 16; ++n) h[n] = 0.0f;
    for (int c = 0; c < SCAN_NCHUNK; ++c) {
        const size_t o = (size_t)((c * 4 + b) * 16) * 2048 + d;
#pragma unroll
        for (int n = 0; n < 16; ++n) {
            const size_t i = o + (size_t)n * 2048;
            Hin[i] = h[n];
            h[n] = Pin[i] * h[n] + Hloc[i];
        }
    }
}

// Pass 3: rerun in-chunk recurrence from Hin; emit y fused with +u*D and
// *silu(gate). Writes y f32 for the final GEMM.
__global__ __launch_bounds__(256) void scan_pass3(
    const float* __restrict__ draw, const float* __restrict__ dtB,
    const float* __restrict__ u, const float* __restrict__ xdbl,
    const float* __restrict__ comb, const float* __restrict__ A_log,
    const float* __restrict__ D_param, const float* __restrict__ Hin,
    float* __restrict__ y)
{
    const int d = blockIdx.x * 256 + threadIdx.x;   // 0..2047
    const int c = blockIdx.y;                       // 0..15
    const int b = blockIdx.z;                       // 0..3
    float a[16], h[16];
    const size_t o = (size_t)((c * 4 + b) * 16) * 2048 + d;
#pragma unroll
    for (int n = 0; n < 16; ++n) {
        a[n] = -__expf(A_log[(size_t)d * 16 + n]);
        h[n] = Hin[o + (size_t)n * 2048];
    }
    const float bias = dtB[d];
    const float Dd = D_param[d];
    const size_t base = (size_t)b * 2048;

    for (int s = c * SCAN_CHUNK; s < (c + 1) * SCAN_CHUNK; ++s) {
        const size_t row = base + s;
        float dt   = softplus_f(draw[row * 2048 + d] + bias);
        float ut   = u[row * 2048 + d];
        float gate = comb[row * 4096 + 2048 + d];
        union BC { float4 q[8]; float f[32]; } bc;
        const float4* bp4 = reinterpret_cast<const float4*>(xdbl + row * 2080 + 2048);
#pragma unroll
        for (int q = 0; q < 8; ++q) bc.q[q] = bp4[q];   // f[0..15]=b, f[16..31]=c

        float yt = 0.0f;
#pragma unroll
        for (int n = 0; n < 16; ++n) {
            float e = __expf(dt * a[n]);
            h[n] = e * h[n] + (dt * bc.f[n]) * ut;
            yt += h[n] * bc.f[16 + n];
        }
        yt += ut * Dd;
        float sig = 1.0f / (1.0f + __expf(-gate));
        y[row * 2048 + d] = yt * (gate * sig);
    }
}

// ---------------------------------------------------------------------------

extern "C" void kernel_launch(void* const* d_in, const int* in_sizes, int n_in,
                              void* d_out, int out_size, void* d_ws, size_t ws_size,
                              hipStream_t stream)
{
    const float* x      = (const float*)d_in[0];   // [4,2048,1024]
    const float* inW    = (const float*)d_in[1];   // [1024,4096]
    const float* convW  = (const float*)d_in[2];   // [2048,1,3]
    const float* convB  = (const float*)d_in[3];   // [2048]
    const float* xprojW = (const float*)d_in[4];   // [2048,2080]
    const float* dtW    = (const float*)d_in[5];   // [2048,2048]
    const float* dtB    = (const float*)d_in[6];   // [2048]
    const float* outW   = (const float*)d_in[7];   // [2048,1024]
    const float* Alog   = (const float*)d_in[8];   // [2048,16]
    const float* Dp     = (const float*)d_in[9];   // [2048]
    float* out = (float*)d_out;                    // [4,2048,1024]

    char* ws = (char*)d_ws;
    float* comb  = (float*)(ws);                      // 8192x4096 f32 (128 MB)
    float* u     = (float*)(ws + 134217728ull);       // 8192x2048 f32 ( 64 MB)
    float* xdbl  = (float*)(ws + 201326592ull);       // 8192x2080 f32 ( 65 MB)
    float* draw  = (float*)(ws + 269484032ull);       // 8192x2048 f32 ( 64 MB)
    float* y     = (float*)(ws + 336592896ull);       // 8192x2048 f32 ( 64 MB)
    float* Pbuf  = (float*)(ws + 403701760ull);       // 16x4x16x2048    (  8 MB)
    float* Hloc  = (float*)(ws + 412090368ull);       //                 (  8 MB)
    float* Hin   = (float*)(ws + 420478976ull);       //                 (  8 MB)

    const int M = 8192;

    // combined = x @ in_proj_w
    gemm_bf16_wmma<<<dim3(32, 64), 256, 0, stream>>>(
        x, inW, comb, M, 4096, 1024, 1024, 4096, 4096);
    // u = silu(depthwise_conv(combined[:, :2048]) + conv_b)
    conv_silu_kernel<<<dim3(8, 2048, 4), 256, 0, stream>>>(comb, convW, convB, u);
    // x_dbl = u @ x_proj_w   (N=2080, guarded tail block)
    gemm_bf16_wmma<<<dim3(17, 64), 256, 0, stream>>>(
        u, xprojW, xdbl, M, 2080, 2048, 2048, 2080, 2080);
    // delta_raw = x_dbl[:, :2048] @ dt_proj_w   (lda=2080 reads the slice)
    gemm_bf16_wmma<<<dim3(16, 64), 256, 0, stream>>>(
        xdbl, dtW, draw, M, 2048, 2048, 2080, 2048, 2048);
    // chunked scan (softplus folded into passes 1 & 3)
    scan_pass1<<<dim3(8, SCAN_NCHUNK, 4), 256, 0, stream>>>(
        draw, dtB, u, xdbl, Alog, Pbuf, Hloc);
    scan_combine<<<dim3(32), 256, 0, stream>>>(Pbuf, Hloc, Hin);
    scan_pass3<<<dim3(8, SCAN_NCHUNK, 4), 256, 0, stream>>>(
        draw, dtB, u, xdbl, comb, Alog, Dp, Hin, y);
    // out = y @ out_proj_w
    gemm_bf16_wmma<<<dim3(8, 64), 256, 0, stream>>>(
        y, outW, out, M, 1024, 2048, 2048, 1024, 1024);
}
